// FocalLoss_20804821582285
// MI455X (gfx1250) — matrix operand order repaired and checked by reading, not compiled, not run on hardware
//
#include <hip/hip_runtime.h>
#include <hip/hip_bf16.h>
#include <stdint.h>

typedef __attribute__((ext_vector_type(16))) _Float16 v16h;
typedef __attribute__((ext_vector_type(8)))  float    v8f;

#define N_LOC        21824
#define NUM_CLASSES  80
#define BATCH        16
#define MAX_BOXES    16
#define WAVES_PB     11
#define THREADS      (WAVES_PB * 32)   // 352
#define BLOCKS_X     124               // 124 * 11 waves * 16 locs = 21824 exactly

// ---------------------------------------------------------------------------
// Main kernel: one wave = 16 locations x 80 classes (5 WMMA class-tiles).
// targets tile = min( inside(16x32,f16) @ onehot(32x16,f16), 1 )  via
// v_wmma_f32_16x16x32_f16, then streaming focal loss over classifications.
// ---------------------------------------------------------------------------
__global__ __launch_bounds__(THREADS) void focal_main(
    const float* __restrict__ cls,   // [B, N_LOC, 80]
    const float* __restrict__ ann,   // [B, 16, 5]
    const int*   __restrict__ xg,    // [N_LOC]
    const int*   __restrict__ yg,    // [N_LOC]
    float* __restrict__ partialLoss, // [B, BLOCKS_X]
    float* __restrict__ partialNum)  // [B, BLOCKS_X]
{
  __shared__ float  annL[MAX_BOXES * 5];
  __shared__ float4 boundsL[4 * 16];          // [level][box] = (lox, loy, hix, hiy)
  __shared__ uint4  bfragL[5 * 32 * 2];       // B fragments: [ct][lane] -> 32 bytes
  __shared__ float  redL[WAVES_PB];
  __shared__ float  redN[WAVES_PB];

  const int b   = blockIdx.y;
  const int tid = threadIdx.x;

  // ---- stage annotations ----
  if (tid < MAX_BOXES * 5) annL[tid] = ann[b * MAX_BOXES * 5 + tid];
  __syncthreads();

  // ---- per-(level,box) effective-region bounds (levels 0..3 only) ----
  if (tid < 64) {
    int lvl = tid >> 4, j = tid & 15;
    float s  = (float)(8 << lvl);
    float x1 = annL[j * 5 + 0], y1 = annL[j * 5 + 1];
    float x2 = annL[j * 5 + 2], y2 = annL[j * 5 + 3];
    bool  valid = (annL[j * 5 + 4] != -1.0f);
    float px1 = floorf((x1 + s - 1.0f) / s);
    float py1 = floorf((y1 + s - 1.0f) / s);
    float px2 = floorf((x2 + s - 1.0f) / s);
    float py2 = floorf((y2 + s - 1.0f) / s);
    float pw = px2 - px1, ph = py2 - py1;
    float lox = floorf(px1 + 0.4f * pw + 1.0f);
    float hix = floorf(px2 - 0.4f * pw + 1.0f);
    float loy = floorf(py1 + 0.4f * ph + 1.0f);
    float hiy = floorf(py2 - 0.4f * ph + 1.0f);
    if (!valid) { lox = 3.0e38f; hix = -3.0e38f; loy = 3.0e38f; hiy = -3.0e38f; }
    boundsL[tid] = make_float4(lox, loy, hix, hiy);
  }

  // ---- build 5 one-hot B fragments (32x16 f16, K padded past 16 boxes) ----
  // B layout assumption: lane L holds column n = L%16; element j<8 <-> K = 8*(L/16)+j, j>=8 <-> K = 16+8*(L/16)+(j-8).
  if (tid < 160) {
    int ct = tid / 32, ln = tid % 32;
    int n = ln & 15, bhi = ln >> 4;
    uint32_t w[8] = {0u, 0u, 0u, 0u, 0u, 0u, 0u, 0u};
    {  // K = 0..15 -> real boxes (elements 0..7 of each half); K >= 16 is zero padding
      int target = ct * 16 + n;
      for (int j = 0; j < 8; ++j) {
        int cj = (int)annL[(j + 8 * bhi) * 5 + 4];
        if (cj == target) w[j >> 1] |= 0x3C00u << ((j & 1) * 16);  // fp16 1.0
      }
    }
    bfragL[(ct * 32 + ln) * 2 + 0] = make_uint4(w[0], w[1], w[2], w[3]);
    bfragL[(ct * 32 + ln) * 2 + 1] = make_uint4(w[4], w[5], w[6], w[7]);
  }
  __syncthreads();

  // ---- per-lane identity within the wave tile ----
  const int lane = tid & 31;
  const int wave = tid >> 5;
  const int m16  = lane & 15;
  const int hi   = lane >> 4;
  const int locBase = (blockIdx.x * WAVES_PB + wave) * 16;
  const int locIdx  = locBase + m16;   // always < 21824 (exact tiling)

  int lvl = (locIdx < 16384) ? 0 : (locIdx < 20480) ? 1 :
            (locIdx < 21504) ? 2 : (locIdx < 21760) ? 3 : 4;
  const bool lvlOK = (lvl < 4);
  const int  blvl  = lvlOK ? lvl : 3;
  const float xf = (float)xg[locIdx];
  const float yf = (float)yg[locIdx];

  // ---- build A fragment: row M = m16, elements 0..7 <-> boxes j+8*hi ----
  v16h a;
  #pragma unroll
  for (int j = 0; j < 16; ++j) a[j] = (_Float16)0.0f;
  int anyIns = 0;
  #pragma unroll
  for (int j8 = 0; j8 < 8; ++j8) {
    float4 bb = boundsL[blvl * 16 + (j8 + 8 * hi)];
    bool inside = lvlOK && (xf > bb.x) && (xf < bb.z) && (yf > bb.y) && (yf < bb.w);
    a[j8] = inside ? (_Float16)1.0f : (_Float16)0.0f;
    anyIns |= inside ? 1 : 0;
  }
  int anyFull = anyIns | __shfl_xor(anyIns, 16, 32);       // OR the two box halves
  float numContrib = (hi == 0 && anyFull) ? 1.0f : 0.0f;   // count each loc once

  // ---- streaming focal loss: 5 class tiles x 8 rows per lane ----
  const float* pp = cls + ((size_t)b * N_LOC + locBase + 8 * hi) * NUM_CLASSES + m16;
  float lossAcc = 0.0f;

  #pragma unroll
  for (int ct = 0; ct < 5; ++ct) {
    union { uint4 u[2]; v16h h; } bu;
    bu.u[0] = bfragL[(ct * 32 + lane) * 2 + 0];
    bu.u[1] = bfragL[(ct * 32 + lane) * 2 + 1];
    v8f c = {0.f, 0.f, 0.f, 0.f, 0.f, 0.f, 0.f, 0.f};
    v8f d = __builtin_amdgcn_wmma_f32_16x16x32_f16(
        false, a, false, bu.h, (short)0, c, false, false);
    #pragma unroll
    for (int r = 0; r < 8; ++r) {
      float t  = fminf(d[r], 1.0f);                     // targets = min(einsum, 1)
      float p  = __builtin_nontemporal_load(pp + (size_t)r * NUM_CLASSES + ct * 16);
      float pc = fminf(fmaxf(p, 1.0e-4f), 1.0f - 1.0e-4f);
      bool  pos = (t > 0.5f);
      float av  = pos ? pc : (1.0f - pc);               // arg of the single log
      float q   = 1.0f - av;                            // focal base
      float wgt = pos ? 0.25f : 0.75f;                  // alpha / (1-alpha)
      lossAcc  -= wgt * q * q * __logf(av);
    }
  }

  // ---- deterministic reduction: wave shfl tree -> LDS -> block partial ----
  #pragma unroll
  for (int off = 16; off > 0; off >>= 1) {
    lossAcc    += __shfl_xor(lossAcc,    off, 32);
    numContrib += __shfl_xor(numContrib, off, 32);
  }
  if (lane == 0) { redL[wave] = lossAcc; redN[wave] = numContrib; }
  __syncthreads();
  if (tid == 0) {
    float L = 0.0f, N = 0.0f;
    for (int w = 0; w < WAVES_PB; ++w) { L += redL[w]; N += redN[w]; }
    partialLoss[b * BLOCKS_X + blockIdx.x] = L;
    partialNum [b * BLOCKS_X + blockIdx.x] = N;
  }
}

// ---------------------------------------------------------------------------
// Final: fixed-order per-image reduce, num_pos normalization, batch mean.
// ---------------------------------------------------------------------------
__global__ __launch_bounds__(256) void focal_final(
    const float* __restrict__ partialLoss,
    const float* __restrict__ partialNum,
    const float* __restrict__ ann,
    float* __restrict__ out)
{
  __shared__ float sL[8], sN[8];
  const int tid = threadIdx.x, lane = tid & 31, wave = tid >> 5;
  float acc = 0.0f;
  for (int b = 0; b < BATCH; ++b) {
    float vL = 0.0f, vN = 0.0f;
    if (tid < BLOCKS_X) {
      vL = partialLoss[b * BLOCKS_X + tid];
      vN = partialNum [b * BLOCKS_X + tid];
    }
    #pragma unroll
    for (int off = 16; off > 0; off >>= 1) {
      vL += __shfl_xor(vL, off, 32);
      vN += __shfl_xor(vN, off, 32);
    }
    if (lane == 0) { sL[wave] = vL; sN[wave] = vN; }
    __syncthreads();
    if (tid == 0) {
      float L = 0.0f, N = 0.0f;
      for (int w = 0; w < 8; ++w) { L += sL[w]; N += sN[w]; }
      bool hb = false;
      for (int j = 0; j < MAX_BOXES; ++j)
        hb = hb || (ann[b * MAX_BOXES * 5 + j * 5 + 4] != -1.0f);
      acc += hb ? (L / fmaxf(N, 1.0f)) : 0.0f;
    }
    __syncthreads();
  }
  if (tid == 0) { out[0] = acc / (float)BATCH; out[1] = 0.0f; }
}

extern "C" void kernel_launch(void* const* d_in, const int* in_sizes, int n_in,
                              void* d_out, int out_size, void* d_ws, size_t ws_size,
                              hipStream_t stream) {
  (void)in_sizes; (void)n_in; (void)out_size; (void)ws_size;
  const float* cls = (const float*)d_in[0];  // classifications [16,21824,80]
  const float* ann = (const float*)d_in[2];  // annotations     [16,16,5]
  const int*   xg  = (const int*)  d_in[4];  // x_grid_order    [21824]
  const int*   yg  = (const int*)  d_in[5];  // y_grid_order    [21824]
  float* out = (float*)d_out;                // [cls_loss_mean, reg_loss_mean]

  float* pl = (float*)d_ws;                  // [16, 124]
  float* pn = pl + BATCH * BLOCKS_X;         // [16, 124]

  dim3 grid(BLOCKS_X, BATCH);
  focal_main<<<grid, THREADS, 0, stream>>>(cls, ann, xg, yg, pl, pn);
  focal_final<<<1, 256, 0, stream>>>(pl, pn, ann, out);
}
